// VQGAN_2379411882544
// MI455X (gfx1250) — matrix operand order repaired and checked by reading, not compiled, not run on hardware
//
#include <hip/hip_runtime.h>
#include <hip/hip_bf16.h>

// ---------------------------------------------------------------------------
// Types for WMMA fragments (wave32, v_wmma_f32_16x16x32_bf16)
// ---------------------------------------------------------------------------
typedef __attribute__((ext_vector_type(16))) __bf16 v16bf;
typedef __attribute__((ext_vector_type(8)))  float  v8f;

union AFrag {
    v16bf v;
    uint4 q[2];
    unsigned short u[16];
};

__device__ __forceinline__ unsigned short f32_to_bf16(float f) {
    unsigned int u = __float_as_uint(f);
    u += 0x7fffu + ((u >> 16) & 1u);   // round to nearest even
    return (unsigned short)(u >> 16);
}

// ---------------------------------------------------------------------------
// Weight packing: f32 OIHW -> bf16 [Cout][Kp]  (conv: k = ci*16 + kh*4 + kw)
// ---------------------------------------------------------------------------
__global__ void pack_conv_w(const float* __restrict__ w, unsigned short* __restrict__ out,
                            int Cin, int Cout, int Kp) {
    int t = blockIdx.x * blockDim.x + threadIdx.x;
    int co = t / Kp;
    int k  = t - co * Kp;
    if (co >= Cout) return;
    int ci = k >> 4, r = k & 15;
    float val = (ci < Cin) ? w[((size_t)co * Cin + ci) * 16 + r] : 0.f;
    out[t] = f32_to_bf16(val);
}

// tconv: per parity class pc = ph*2+pw, k = ci*4 + th*2 + tw
// weight element = w[co][ci][ph+2*th][pw+2*tw]; zero-padded to CoutPad rows.
__global__ void pack_tconv_w(const float* __restrict__ w, unsigned short* __restrict__ out,
                             int Cin, int Cout, int CoutPad) {
    int Kp = Cin * 4;
    int t = blockIdx.x * blockDim.x + threadIdx.x;
    int perClass = CoutPad * Kp;
    int pc = t / perClass;
    int r  = t - pc * perClass;
    if (pc >= 4) return;
    int co = r / Kp;
    int k  = r - co * Kp;
    int ci = k >> 2, th = (k >> 1) & 1, tw = k & 1;
    int ph = pc >> 1, pw = pc & 1;
    float val = 0.f;
    if (co < Cout)
        val = w[(((size_t)co * Cin + ci) * 4 + (ph + 2 * th)) * 4 + (pw + 2 * tw)];
    out[t] = f32_to_bf16(val);
}

// ---------------------------------------------------------------------------
// Codebook: bf16 pack + per-row squared norm
// ---------------------------------------------------------------------------
__global__ void vq_pack_codebook(const float* __restrict__ cb,
                                 unsigned short* __restrict__ cbf,
                                 float* __restrict__ cnorm) {
    int r = blockIdx.x;            // 1024 rows
    int d = threadIdx.x;           // 256
    float v = cb[(size_t)r * 256 + d];
    cbf[(size_t)r * 256 + d] = f32_to_bf16(v);
    __shared__ float red[256];
    red[d] = v * v;
    __syncthreads();
    for (int s = 128; s > 0; s >>= 1) {
        if (d < s) red[d] += red[d + s];
        __syncthreads();
    }
    if (d == 0) cnorm[r] = red[0];
}

// ---------------------------------------------------------------------------
// Implicit-GEMM conv / tconv.  MODE 0: conv 4x4 s2 p1.  MODE 1: tconv 4x4 s2
// SAME decomposed into 4 parity classes (blockIdx.z).
// Block = 128 threads = 4 waves. Each block: 16 output pixels x 64 channels.
// A tile (16x32 bf16) staged in LDS, shared by all 4 waves.
// ---------------------------------------------------------------------------
template <int MODE, int RELU>
__global__ __launch_bounds__(128) void conv_gemm_n64(
    const float* __restrict__ in, const unsigned short* __restrict__ wb,
    const float* __restrict__ bias, float* __restrict__ out,
    int Cin, int Hin, int Win, int Cout, int Hout, int Wout, int Kp) {

    __shared__ alignas(16) unsigned short lds_a[16 * 32];

    int tid  = threadIdx.x;
    int lane = tid & 31;
    int wave = tid >> 5;
    int mtile = blockIdx.x;

    int ph = 0, pw = 0;
    const unsigned short* wbp = wb;
    int Ho = Hout, Wo = Wout;
    if (MODE == 1) {
        int pc = blockIdx.z;
        ph = pc >> 1; pw = pc & 1;
        wbp = wb + (size_t)pc * Cout * Kp;   // per-class packed weights
        Ho = Hout >> 1; Wo = Wout >> 1;      // per-class output grid
    }
    int HoWo = Ho * Wo;

    int n  = lane & 15;
    int co = blockIdx.y * 64 + wave * 16 + n;
    int kbase_lane = (lane < 16) ? 0 : 16;
    const unsigned short* bp = wbp + (size_t)co * Kp + kbase_lane;

    v8f acc = {};

    for (int k0 = 0; k0 < Kp; k0 += 32) {
        // cooperative im2col gather of the 16x32 A tile into LDS
        for (int i = tid; i < 512; i += 128) {
            int mloc = i >> 5;
            int kk   = i & 31;
            int k    = k0 + kk;
            int m    = mtile * 16 + mloc;
            int b  = m / HoWo;
            int r  = m - b * HoWo;
            int oy = r / Wo;
            int ox = r - oy * Wo;
            float val = 0.f;
            if (MODE == 0) {
                int ci = k >> 4, t = k & 15, kh = t >> 2, kw = t & 3;
                int iy = oy * 2 - 1 + kh, ix = ox * 2 - 1 + kw;
                if (ci < Cin && (unsigned)iy < (unsigned)Hin && (unsigned)ix < (unsigned)Win)
                    val = in[(((size_t)b * Cin + ci) * Hin + iy) * Win + ix];
            } else {
                int ci = k >> 2, th = (k >> 1) & 1, tw = k & 1;
                int iy = oy + ph + th - 1, ix = ox + pw + tw - 1;
                if ((unsigned)iy < (unsigned)Hin && (unsigned)ix < (unsigned)Win)
                    val = in[(((size_t)b * Cin + ci) * Hin + iy) * Win + ix];
            }
            lds_a[i] = f32_to_bf16(val);
        }
        __syncthreads();

        // A fragment: lanes 0-15 row m, K 0-15; lanes 16-31 K 16-31
        AFrag a;
        int mloc = lane & 15;
        int off  = (lane < 16) ? 0 : 16;
        const unsigned short* lp = lds_a + mloc * 32;
        a.q[0] = *(const uint4*)(lp + off);
        a.q[1] = *(const uint4*)(lp + off + 8);

        // B fragment: 16 contiguous bf16 per lane from packed weights
        v16bf bv = *(const v16bf*)(bp + k0);

        acc = __builtin_amdgcn_wmma_f32_16x16x32_bf16(false, a.v, false, bv,
                                                      (short)0, acc, false, false);
        __syncthreads();
    }

    // epilogue: bias (+relu), scatter to NCHW
    float bsv = bias[co];
#pragma unroll
    for (int v = 0; v < 8; ++v) {
        int mloc = v + ((lane < 16) ? 0 : 8);
        int m  = mtile * 16 + mloc;
        int b  = m / HoWo;
        int r  = m - b * HoWo;
        int oy = r / Wo;
        int ox = r - oy * Wo;
        if (MODE == 1) { oy = oy * 2 + ph; ox = ox * 2 + pw; }
        float val = acc[v] + bsv;
        if (RELU) val = fmaxf(val, 0.f);
        out[(((size_t)b * Cout + co) * Hout + oy) * Wout + ox] = val;
    }
}

// Single-wave variant for tiny Cout (final tconv, Cout=3, packed to CoutPad=16).
template <int MODE, int RELU>
__global__ __launch_bounds__(32) void conv_gemm_n16(
    const float* __restrict__ in, const unsigned short* __restrict__ wb,
    const float* __restrict__ bias, float* __restrict__ out,
    int Cin, int Hin, int Win, int Cout, int CoutPad, int Hout, int Wout, int Kp) {

    int lane  = threadIdx.x;
    int mtile = blockIdx.x;

    int ph = 0, pw = 0;
    const unsigned short* wbp = wb;
    int Ho = Hout, Wo = Wout;
    if (MODE == 1) {
        int pc = blockIdx.z;
        ph = pc >> 1; pw = pc & 1;
        wbp = wb + (size_t)pc * CoutPad * Kp;
        Ho = Hout >> 1; Wo = Wout >> 1;
    }
    int HoWo = Ho * Wo;

    int n  = lane & 15;
    int co = blockIdx.y * 16 + n;
    const unsigned short* bp = wbp + (size_t)co * Kp + ((lane < 16) ? 0 : 16);

    int mloc = lane & 15;
    int m    = mtile * 16 + mloc;
    int b  = m / HoWo;
    int r  = m - b * HoWo;
    int oy = r / Wo;
    int ox = r - oy * Wo;
    int off = (lane < 16) ? 0 : 16;

    v8f acc = {};
    for (int k0 = 0; k0 < Kp; k0 += 32) {
        AFrag a;
#pragma unroll
        for (int e = 0; e < 16; ++e) {
            int k = k0 + off + e;
            float val = 0.f;
            if (MODE == 0) {
                int ci = k >> 4, t = k & 15, kh = t >> 2, kw = t & 3;
                int iy = oy * 2 - 1 + kh, ix = ox * 2 - 1 + kw;
                if (ci < Cin && (unsigned)iy < (unsigned)Hin && (unsigned)ix < (unsigned)Win)
                    val = in[(((size_t)b * Cin + ci) * Hin + iy) * Win + ix];
            } else {
                int ci = k >> 2, th = (k >> 1) & 1, tw = k & 1;
                int iy = oy + ph + th - 1, ix = ox + pw + tw - 1;
                if ((unsigned)iy < (unsigned)Hin && (unsigned)ix < (unsigned)Win)
                    val = in[(((size_t)b * Cin + ci) * Hin + iy) * Win + ix];
            }
            a.u[e] = f32_to_bf16(val);
        }
        v16bf bv = *(const v16bf*)(bp + k0);
        acc = __builtin_amdgcn_wmma_f32_16x16x32_bf16(false, a.v, false, bv,
                                                      (short)0, acc, false, false);
    }

    if (co < Cout) {
        float bsv = bias[co];
#pragma unroll
        for (int v = 0; v < 8; ++v) {
            int ml = v + ((lane < 16) ? 0 : 8);
            int mm = mtile * 16 + ml;
            int bb = mm / HoWo;
            int rr = mm - bb * HoWo;
            int yy = rr / Wo;
            int xx = rr - yy * Wo;
            if (MODE == 1) { yy = yy * 2 + ph; xx = xx * 2 + pw; }
            float val = acc[v] + bsv;
            if (RELU) val = fmaxf(val, 0.f);
            out[(((size_t)bb * Cout + co) * Hout + yy) * Wout + xx] = val;
        }
    }
}

// ---------------------------------------------------------------------------
// VQ argmin: one wave handles 16 z-vectors; sweep 1024 codes with WMMA.
// argmin(|z|^2 - 2 z.c + |c|^2) == argmin(|c|^2 - 2 z.c)  (|z|^2 const per row)
// z_e layout: [8][256][32*32] NCHW -> z[m][d] at ze[(b*256+d)*1024 + s]
// ---------------------------------------------------------------------------
__global__ __launch_bounds__(32) void vq_argmin_kernel(
    const float* __restrict__ ze, const unsigned short* __restrict__ cbf,
    const float* __restrict__ cnorm, int* __restrict__ idx) {

    int lane  = threadIdx.x;
    int mtile = blockIdx.x;                 // 512 tiles of 16 rows
    int mloc  = lane & 15;
    int m = mtile * 16 + mloc;
    int b = m >> 10;
    int s = m & 1023;
    const float* zp = ze + (size_t)b * 256 * 1024 + s;
    int off = (lane < 16) ? 0 : 16;

    // preload the 16x256 A fragments (8 k-steps) into registers
    AFrag af[8];
#pragma unroll
    for (int ks = 0; ks < 8; ++ks) {
        int kb = ks * 32;
#pragma unroll
        for (int e = 0; e < 8; ++e)
            af[ks].u[e] = f32_to_bf16(zp[(size_t)(kb + off + e) * 1024]);
#pragma unroll
        for (int e = 0; e < 8; ++e)
            af[ks].u[8 + e] = f32_to_bf16(zp[(size_t)(kb + off + 8 + e) * 1024]);
    }

    float minv[8];
    int   mini[8];
#pragma unroll
    for (int v = 0; v < 8; ++v) { minv[v] = 3.4e38f; mini[v] = 0; }

    int n = lane & 15;
    int kb2 = (lane < 16) ? 0 : 16;
    for (int n0 = 0; n0 < 1024; n0 += 16) {
        v8f acc = {};
        const unsigned short* bp = cbf + (size_t)(n0 + n) * 256 + kb2;
#pragma unroll
        for (int ks = 0; ks < 8; ++ks) {
            v16bf bv = *(const v16bf*)(bp + ks * 32);
            acc = __builtin_amdgcn_wmma_f32_16x16x32_bf16(false, af[ks].v, false, bv,
                                                          (short)0, acc, false, false);
        }
        float cn = cnorm[n0 + n];
#pragma unroll
        for (int v = 0; v < 8; ++v) {
            float d = cn - 2.f * acc[v];
            if (d < minv[v]) { minv[v] = d; mini[v] = n0 + n; }
        }
    }

    // argmin reduce across the 16 lanes sharing each row (ties -> lower index)
#pragma unroll
    for (int v = 0; v < 8; ++v) {
        float mv = minv[v];
        int   mi = mini[v];
        for (int sft = 8; sft >= 1; sft >>= 1) {
            float ov = __shfl_xor(mv, sft);
            int   oi = __shfl_xor(mi, sft);
            if (ov < mv || (ov == mv && oi < mi)) { mv = ov; mi = oi; }
        }
        if ((lane & 15) == 0) {
            int ml = v + ((lane < 16) ? 0 : 8);
            idx[mtile * 16 + ml] = mi;
        }
    }
}

// z_q gather (straight-through forward value) + vq_loss accumulation.
// vq_loss = (1 + BETA) * mean((z - q)^2),  BETA = 0.25
__global__ void vq_gather_loss(const float* __restrict__ ze, const float* __restrict__ cb,
                               const int* __restrict__ idx, float* __restrict__ zq,
                               float* __restrict__ loss_out) {
    int m = blockIdx.x;      // 8192
    int d = threadIdx.x;     // 256
    int b = m >> 10;
    int s = m & 1023;
    int id = idx[m];
    float q = cb[(size_t)id * 256 + d];
    float z = ze[((size_t)b * 256 + d) * 1024 + s];
    zq[((size_t)b * 256 + d) * 1024 + s] = q;
    float diff = z - q;
    __shared__ float red[256];
    red[d] = diff * diff;
    __syncthreads();
    for (int sft = 128; sft > 0; sft >>= 1) {
        if (d < sft) red[d] += red[d + sft];
        __syncthreads();
    }
    if (d == 0) atomicAdd(loss_out, red[0] * (1.25f / (8192.f * 256.f)));
}

// ---------------------------------------------------------------------------
// Launch
// ---------------------------------------------------------------------------
extern "C" void kernel_launch(void* const* d_in, const int* in_sizes, int n_in,
                              void* d_out, int out_size, void* d_ws, size_t ws_size,
                              hipStream_t stream) {
    (void)in_sizes; (void)n_in; (void)out_size;

    const float* x   = (const float*)d_in[0];
    const float* ew1 = (const float*)d_in[1];
    const float* eb1 = (const float*)d_in[2];
    const float* ew2 = (const float*)d_in[3];
    const float* eb2 = (const float*)d_in[4];
    const float* ew3 = (const float*)d_in[5];
    const float* eb3 = (const float*)d_in[6];
    const float* cbk = (const float*)d_in[7];
    const float* dw1 = (const float*)d_in[8];
    const float* db1 = (const float*)d_in[9];
    const float* dw2 = (const float*)d_in[10];
    const float* db2 = (const float*)d_in[11];
    const float* dw3 = (const float*)d_in[12];
    const float* db3 = (const float*)d_in[13];

    char* base = (char*)d_ws;
    size_t off = 0;
    auto take = [&](size_t nbytes) -> void* {
        void* p = base + off;
        off += (nbytes + 255) & ~(size_t)255;
        return p;
    };
    unsigned short* wq1  = (unsigned short*)take((size_t)128 * 64 * 2);        // enc1 packed
    unsigned short* wq2  = (unsigned short*)take((size_t)256 * 2048 * 2);      // enc2
    unsigned short* wq3  = (unsigned short*)take((size_t)256 * 4096 * 2);      // enc3
    unsigned short* wd1p = (unsigned short*)take((size_t)4 * 256 * 1024 * 2);  // dec1 per-class
    unsigned short* wd2p = (unsigned short*)take((size_t)4 * 128 * 1024 * 2);  // dec2
    unsigned short* wd3p = (unsigned short*)take((size_t)4 * 16 * 512 * 2);    // dec3 (padded)
    unsigned short* cbf  = (unsigned short*)take((size_t)1024 * 256 * 2);
    float* cnorm = (float*)take(1024 * 4);
    int*   idxb  = (int*)take(8192 * 4);
    float* h1 = (float*)take((size_t)16777216 * 4);  // [8,128,128,128]
    float* h2 = (float*)take((size_t)8388608 * 4);   // [8,256,64,64]
    float* ze = (float*)take((size_t)2097152 * 4);   // [8,256,32,32]
    float* zq = (float*)take((size_t)2097152 * 4);
    float* g1 = (float*)take((size_t)8388608 * 4);   // [8,256,64,64]
    float* g2 = (float*)take((size_t)16777216 * 4);  // [8,128,128,128]
    if (off > ws_size) return;

    float* xhat = (float*)d_out;                                  // 1572864 floats
    float* loss = (float*)((char*)d_out + (size_t)1572864 * 4);   // scalar
    hipMemsetAsync(loss, 0, 4, stream);

    // --- weight / codebook packing (bf16) ---
    pack_conv_w <<<(128 * 64) / 256, 256, 0, stream>>>(ew1, wq1, 3, 128, 64);
    pack_conv_w <<<(256 * 2048) / 256, 256, 0, stream>>>(ew2, wq2, 128, 256, 2048);
    pack_conv_w <<<(256 * 4096) / 256, 256, 0, stream>>>(ew3, wq3, 256, 256, 4096);
    pack_tconv_w<<<(4 * 256 * 1024) / 256, 256, 0, stream>>>(dw1, wd1p, 256, 256, 256);
    pack_tconv_w<<<(4 * 128 * 1024) / 256, 256, 0, stream>>>(dw2, wd2p, 256, 128, 128);
    pack_tconv_w<<<(4 * 16 * 512) / 256, 256, 0, stream>>>(dw3, wd3p, 128, 3, 16);
    vq_pack_codebook<<<1024, 256, 0, stream>>>(cbk, cbf, cnorm);

    // --- encoder ---
    conv_gemm_n64<0, 1><<<dim3(8192, 2, 1), 128, 0, stream>>>(x,  wq1, eb1, h1, 3,   256, 256, 128, 128, 128, 64);
    conv_gemm_n64<0, 1><<<dim3(2048, 4, 1), 128, 0, stream>>>(h1, wq2, eb2, h2, 128, 128, 128, 256, 64,  64,  2048);
    conv_gemm_n64<0, 0><<<dim3(512,  4, 1), 128, 0, stream>>>(h2, wq3, eb3, ze, 256, 64,  64,  256, 32,  32,  4096);

    // --- vector quantization ---
    vq_argmin_kernel<<<512, 32, 0, stream>>>(ze, cbf, cnorm, idxb);
    vq_gather_loss<<<8192, 256, 0, stream>>>(ze, cbk, idxb, zq, loss);

    // --- decoder (tconv as 4 parity-class GEMMs) ---
    conv_gemm_n64<1, 1><<<dim3(512,  4, 4), 128, 0, stream>>>(zq, wd1p, db1, g1, 256, 32,  32,  256, 64,  64,  1024);
    conv_gemm_n64<1, 1><<<dim3(2048, 2, 4), 128, 0, stream>>>(g1, wd2p, db2, g2, 256, 64,  64,  128, 128, 128, 1024);
    conv_gemm_n16<1, 0><<<dim3(8192, 1, 4), 32,  0, stream>>>(g2, wd3p, db3, xhat, 128, 128, 128, 3, 16, 256, 256, 512);
}
